// SparseMoEBlock__9328668967108
// MI455X (gfx1250) — compile-verified, matrix operand order checked by reading, not run on hardware
//
#include <hip/hip_runtime.h>

// ---------------- problem dims ----------------
#define S_TOK 4096   // B*SEQ
#define DIM   1024   // D
#define NEXP  8      // E
#define FDIM  4096   // F
#define KTOP  8192   // S * CAPACITY

// ---------------- workspace layout (bytes, 64-aligned) ----------------
#define OFF_SCORES 0                       // float[E*S]            = 131072 B
#define OFF_COUNTS 131072                  // int[E]
#define OFF_EOFF   131136                  // int[E+1]
#define OFF_TOK    131584                  // int[KTOP]             = 32768 B
#define OFF_GATE   164352                  // float[KTOP]           = 32768 B
#define OFF_XB     197376                  // bf16[S*D]             = 8 MB
#define OFF_HS     8585984                 // bf16[S*F]             = 32 MB
#define OFF_HX     42140416                // bf16[KTOP*F]          = 64 MB
// total ~ 109,249,280 B

// ---------------- WMMA types ----------------
typedef __bf16 v16bf __attribute__((ext_vector_type(16)));
typedef float  v8f   __attribute__((ext_vector_type(8)));

union FragBF { unsigned u[8]; v16bf v; };

// fp32 -> bf16 (round-to-nearest-even) raw bits.
__device__ __forceinline__ unsigned short f2bf(float f) {
  unsigned u = __float_as_uint(f);
  u = u + 0x7FFFu + ((u >> 16) & 1u);
  return (unsigned short)(u >> 16);
}

__device__ __forceinline__ float gelu_tanh(float x) {
  float x3 = x * x * x;
  float u = 0.7978845608028654f * (x + 0.044715f * x3);
  return 0.5f * x * (1.0f + tanhf(u));
}

// 16xK=32 bf16 fragment from LDS tile stored row-major [row][k] (ld even).
// ISA 16-bit A layout: lane = row + 16*g ; per VGPR v the packed K pair is
// k0 = 2*(v&3) + 8*g + 16*(v>>2). B uses same map with row := N ([n][k] tile).
__device__ __forceinline__ v16bf load_frag(const unsigned short* base, int lane,
                                           int roff, int ld) {
  FragBF fr;
  const int r = (lane & 15) + roff;
  const int g = lane >> 4;
  const unsigned short* p = base + r * ld;
#pragma unroll
  for (int v = 0; v < 8; ++v) {
    const int k0 = ((v & 3) << 1) + (g << 3) + ((v >> 2) << 4);
    fr.u[v] = *(const unsigned*)(p + k0);
  }
  return fr.v;
}

__device__ __forceinline__ v8f wmma_bf16(v16bf a, v16bf b, v8f c) {
  return __builtin_amdgcn_wmma_f32_16x16x32_bf16(false, a, false, b, (short)0, c,
                                                 false, false);
}

// ---------------- k0: gating logits + softmax (one wave32 / token) --------
__global__ __launch_bounds__(256) void gate_kernel(
    const float* __restrict__ x, const float* __restrict__ gw,
    float* __restrict__ scores) {
  const int lane = threadIdx.x & 31;
  const int t = blockIdx.x * 8 + (threadIdx.x >> 5);
  const float* xr = x + (size_t)t * DIM;
  float acc[NEXP];
#pragma unroll
  for (int e = 0; e < NEXP; ++e) acc[e] = 0.0f;
  for (int d = lane; d < DIM; d += 32) {
    const float xv = xr[d];
#pragma unroll
    for (int e = 0; e < NEXP; ++e) acc[e] += xv * gw[e * DIM + d];
  }
#pragma unroll
  for (int e = 0; e < NEXP; ++e) {
#pragma unroll
    for (int off = 16; off > 0; off >>= 1) acc[e] += __shfl_xor(acc[e], off, 32);
  }
  float m = acc[0];
#pragma unroll
  for (int e = 1; e < NEXP; ++e) m = fmaxf(m, acc[e]);
  float p[NEXP], se = 0.0f;
#pragma unroll
  for (int e = 0; e < NEXP; ++e) { p[e] = __expf(acc[e] - m); se += p[e]; }
  const float inv = 1.0f / se;
  if (lane < NEXP) scores[lane * S_TOK + t] = p[lane] * inv;
}

// ---------------- k2: x fp32 -> bf16, vectorized ----------------
__global__ __launch_bounds__(256) void cvt_kernel(const float* __restrict__ x,
                                                  unsigned short* __restrict__ xb) {
  const int i = (blockIdx.x * 256 + threadIdx.x) * 4;
  const float4 v = *(const float4*)(x + i);
  ushort4 r;
  r.x = f2bf(v.x); r.y = f2bf(v.y); r.z = f2bf(v.z); r.w = f2bf(v.w);
  *(ushort4*)(xb + i) = r;
}

// ---------------- k1: global top-k via single-block LDS bitonic sort -------
// 32768 u64 keys = 256KB dynamic LDS (CDNA5: 320KB/WGP).
__global__ void topk_kernel(const float* __restrict__ scores,
                            int* __restrict__ tok, float* __restrict__ gate,
                            int* __restrict__ eoff, int* __restrict__ counts) {
  extern __shared__ unsigned long long sk[];
  const int Ntot = NEXP * S_TOK;  // 32768
  const int T = blockDim.x;       // 1024
  const int tid = threadIdx.x;
  for (int i = tid; i < Ntot; i += T)
    sk[i] = ((unsigned long long)__float_as_uint(scores[i]) << 32) | (unsigned)i;
  __syncthreads();
  for (int k = 2; k <= Ntot; k <<= 1) {
    for (int j = k >> 1; j > 0; j >>= 1) {
      for (int i = tid; i < Ntot; i += T) {
        const int ixj = i ^ j;
        if (ixj > i) {
          const unsigned long long a = sk[i], b = sk[ixj];
          const bool desc = ((i & k) == 0);
          if (desc ? (a < b) : (a > b)) { sk[i] = b; sk[ixj] = a; }
        }
      }
      __syncthreads();
    }
  }
  __shared__ int s_cnt[NEXP], s_off[NEXP + 1], s_fill[NEXP];
  if (tid < NEXP) s_cnt[tid] = 0;
  __syncthreads();
  for (int i = tid; i < KTOP; i += T) {
    const unsigned idx = (unsigned)sk[i];
    atomicAdd(&s_cnt[idx >> 12], 1);  // S_TOK == 1<<12
  }
  __syncthreads();
  if (tid == 0) {
    int a = 0;
    for (int e = 0; e < NEXP; ++e) { s_off[e] = a; a += s_cnt[e]; }
    s_off[NEXP] = a;
  }
  if (tid < NEXP) s_fill[tid] = 0;
  __syncthreads();
  if (tid < NEXP) { counts[tid] = s_cnt[tid]; eoff[tid] = s_off[tid]; }
  if (tid == 0) eoff[NEXP] = s_off[NEXP];
  for (int i = tid; i < KTOP; i += T) {
    const unsigned long long kv = sk[i];
    const unsigned idx = (unsigned)kv;
    const int e = idx >> 12;
    const int t = idx & (S_TOK - 1);
    const int slot = atomicAdd(&s_fill[e], 1);
    const int pos = s_off[e] + slot;
    tok[pos] = t;
    gate[pos] = __uint_as_float((unsigned)(kv >> 32));
  }
}

// ---------------- fused GEMM core ----------------
// Block tile 64(M) x 128(N), 256 threads = 8 waves (2 in M x 4 in N); each
// wave computes a 2x2 grid of 16x16 WMMA tiles (4 v_wmma per K-step with A/B
// fragment reuse). Double-buffered LDS, software-pipelined staging: one
// barrier per K-step of 32.
// MODE 0: H = gelu(A @ W + b) stored bf16 (A rows gathered by token list, or
//         identity when tok==nullptr).
// MODE 1: Y = A @ W + b; gated atomic scatter to out (expert) or plain store
//         (shared, tok==nullptr). A rows are compacted slots (obase + row).
template <int MODE>
__global__ __launch_bounds__(256) void gemm_kernel(
    const unsigned short* __restrict__ Abf, const float* __restrict__ W,
    const float* __restrict__ bias, const int* __restrict__ tok,
    const int* __restrict__ eoff, const int* __restrict__ counts,
    const float* __restrict__ gate, unsigned short* __restrict__ Hout,
    float* __restrict__ out, int K, int N) {
  const int e = blockIdx.z;
  const int rowBase = blockIdx.y * 64;
  const int nBase = blockIdx.x * 128;
  int cnt, obase;
  if (tok) {
    cnt = counts[e];
    if (rowBase >= cnt) return;
    obase = eoff[e];
  } else {
    cnt = S_TOK;
    obase = 0;
  }
  const float* We = W + (size_t)e * K * N;
  const float* be = bias + (size_t)e * N;

  __shared__ alignas(16) unsigned short sA[2][64 * 32];   // [m][k]
  __shared__ alignas(16) unsigned short sB[2][128 * 34];  // [n][k] padded

  const int tid = threadIdx.x;
  // ---- A staging: thread -> (row r, 8 bf16 at col c), one b128 per step
  const int ar = tid >> 2;
  const int ac = (tid & 3) << 3;
  const int arl = min(rowBase + ar, cnt - 1);  // clamp keeps loads in-bounds
  size_t a_row;
  if (MODE == 0)
    a_row = (size_t)(tok ? tok[obase + arl] : arl);
  else
    a_row = (size_t)(obase + arl);
  const unsigned short* aptr = Abf + a_row * K + ac;
  // ---- B staging: 128 lanes across N (coalesced), 2 groups x 16 K-rows
  const int bn = tid & 127;
  const int bk0 = (tid >> 7) << 4;
  const float* bptr = We + nBase + bn;

  const int lane = tid & 31;
  const int wv = tid >> 5;
  const int mW = (wv & 1) << 5;   // wave M offset (0/32)
  const int nW = (wv >> 1) << 5;  // wave N offset (0/32/64/96)

  v8f acc00 = {0.f, 0.f, 0.f, 0.f, 0.f, 0.f, 0.f, 0.f};
  v8f acc01 = acc00, acc10 = acc00, acc11 = acc00;

  uint4 areg;
  float breg[16];

  // prologue: stage K-step 0 into buffer 0
  areg = *(const uint4*)(aptr);
#pragma unroll
  for (int i = 0; i < 16; ++i) breg[i] = bptr[(size_t)(bk0 + i) * N];
  *(uint4*)(&sA[0][ar * 32 + ac]) = areg;
#pragma unroll
  for (int i = 0; i < 16; ++i) sB[0][bn * 34 + bk0 + i] = f2bf(breg[i]);
  __syncthreads();

  int buf = 0;
  for (int kb = 0; kb < K; kb += 32) {
    const bool has_next = (kb + 32) < K;
    if (has_next) {  // issue next tile's global loads early
      areg = *(const uint4*)(aptr + kb + 32);
#pragma unroll
      for (int i = 0; i < 16; ++i)
        breg[i] = bptr[(size_t)(kb + 32 + bk0 + i) * N];
      if (kb + 64 < K)
        __builtin_prefetch(bptr + (size_t)(kb + 64 + bk0) * N, 0, 1);
    }
    // compute on current buffer: 2 A frags, 2 B frags, 4 wmma
    {
      const unsigned short* pa = sA[buf];
      const unsigned short* pb = sB[buf];
      const v16bf a0 = load_frag(pa, lane, mW, 32);
      const v16bf a1 = load_frag(pa, lane, mW + 16, 32);
      const v16bf b0 = load_frag(pb, lane, nW, 34);
      const v16bf b1 = load_frag(pb, lane, nW + 16, 34);
      acc00 = wmma_bf16(a0, b0, acc00);
      acc01 = wmma_bf16(a0, b1, acc01);
      acc10 = wmma_bf16(a1, b0, acc10);
      acc11 = wmma_bf16(a1, b1, acc11);
    }
    if (has_next) {  // write next tile into the other buffer
      const int nb = buf ^ 1;
      *(uint4*)(&sA[nb][ar * 32 + ac]) = areg;
#pragma unroll
      for (int i = 0; i < 16; ++i) sB[nb][bn * 34 + bk0 + i] = f2bf(breg[i]);
    }
    __syncthreads();
    buf ^= 1;
  }

  // ---- epilogue: C/D layout M = j + 8*(lane>>4), N = lane&15
#pragma unroll
  for (int mi = 0; mi < 2; ++mi) {
#pragma unroll
    for (int ni = 0; ni < 2; ++ni) {
      const v8f a = (mi == 0) ? (ni == 0 ? acc00 : acc01)
                              : (ni == 0 ? acc10 : acc11);
#pragma unroll
      for (int j = 0; j < 8; ++j) {
        const int m = mW + (mi << 4) + j + ((lane >> 4) << 3);
        const int n = nBase + nW + (ni << 4) + (lane & 15);
        const int row = rowBase + m;
        if (row < cnt) {
          const float y = a[j] + be[n];
          if (MODE == 0) {
            Hout[(size_t)(obase + row) * N + n] = f2bf(gelu_tanh(y));
          } else if (tok) {
            const int t = tok[obase + row];
            const float g = gate[obase + row];
            atomicAdd(&out[(size_t)t * N + n], g * y);
          } else {
            out[(size_t)row * N + n] = y;  // shared path fully initializes out
          }
        }
      }
    }
  }
}

// ---------------- host launcher ----------------
extern "C" void kernel_launch(void* const* d_in, const int* in_sizes, int n_in,
                              void* d_out, int out_size, void* d_ws,
                              size_t ws_size, hipStream_t stream) {
  (void)in_sizes; (void)n_in; (void)out_size; (void)ws_size;
  const float* x   = (const float*)d_in[0];
  const float* gw  = (const float*)d_in[1];
  const float* W1  = (const float*)d_in[2];
  const float* b1  = (const float*)d_in[3];
  const float* W2  = (const float*)d_in[4];
  const float* b2  = (const float*)d_in[5];
  const float* Ws1 = (const float*)d_in[6];
  const float* bs1 = (const float*)d_in[7];
  const float* Ws2 = (const float*)d_in[8];
  const float* bs2 = (const float*)d_in[9];
  float* out = (float*)d_out;

  char* ws = (char*)d_ws;
  float*          scores = (float*)(ws + OFF_SCORES);
  int*            counts = (int*)(ws + OFF_COUNTS);
  int*            eoff   = (int*)(ws + OFF_EOFF);
  int*            tok    = (int*)(ws + OFF_TOK);
  float*          gate   = (float*)(ws + OFF_GATE);
  unsigned short* xb     = (unsigned short*)(ws + OFF_XB);
  unsigned short* Hs     = (unsigned short*)(ws + OFF_HS);
  unsigned short* Hx     = (unsigned short*)(ws + OFF_HX);

  // 1) gating scores
  gate_kernel<<<S_TOK / 8, 256, 0, stream>>>(x, gw, scores);
  // 2) bf16 activations
  cvt_kernel<<<(S_TOK * DIM) / 1024, 256, 0, stream>>>(x, xb);
  // 3) global top-k + per-expert compaction (256KB LDS bitonic sort)
  topk_kernel<<<1, 1024, (size_t)(NEXP * S_TOK) * 8, stream>>>(scores, tok, gate,
                                                               eoff, counts);
  // 4) shared-expert path (writes out fully)
  gemm_kernel<0><<<dim3(FDIM / 128, S_TOK / 64, 1), 256, 0, stream>>>(
      xb, Ws1, bs1, nullptr, nullptr, nullptr, nullptr, Hs, nullptr, DIM, FDIM);
  gemm_kernel<1><<<dim3(DIM / 128, S_TOK / 64, 1), 256, 0, stream>>>(
      Hs, Ws2, bs2, nullptr, nullptr, nullptr, nullptr, nullptr, out, FDIM, DIM);
  // 5) sparse expert path (accumulates gated outputs on top)
  gemm_kernel<0><<<dim3(FDIM / 128, S_TOK / 64, NEXP), 256, 0, stream>>>(
      xb, W1, b1, tok, eoff, counts, nullptr, Hx, nullptr, DIM, FDIM);
  gemm_kernel<1><<<dim3(DIM / 128, S_TOK / 64, NEXP), 256, 0, stream>>>(
      Hx, W2, b2, tok, eoff, counts, gate, nullptr, out, FDIM, DIM);
}
